// LoRAConvPerSubject_21268678050413
// MI455X (gfx1250) — matrix operand order repaired and checked by reading, not compiled, not run on hardware
//
#include <hip/hip_runtime.h>

typedef __attribute__((ext_vector_type(16))) __bf16 v16bf;
typedef __attribute__((ext_vector_type(8)))  float  v8f;

// Problem constants (from reference setup_inputs)
#define BB 32
#define CC 64
#define HH 64
#define WW 256
#define OO 64
#define EE 4
#define RR 4
// GEMM tiling
#define TILE_W 128          // spatial columns per workgroup
#define STG_COLS 130        // staged columns (with halo)
#define STG_PITCH 132       // padded column pitch in LDS
// Weight fragment sizes: per (e,t,mt,kc): 32 lanes x 16 halves = 512 bf16
#define FRAG_ELEMS 512
#define FRAGS_PER_E (9 * 4 * 2)

__device__ __forceinline__ unsigned short f32_to_bf16(float f) {
    unsigned int u = __float_as_uint(f);
    u += 0x7FFFu + ((u >> 16) & 1u);   // round-to-nearest-even
    return (unsigned short)(u >> 16);
}

// ---------------------------------------------------------------------------
// Kernel 1: build W_eff[e] = W_conv + 0.25 * sum_r W_B[e,:,r] (x) W_A[e,r]
// and store it pre-swizzled as WMMA A fragments in this code's contiguous k order (matching B):
//   fragment (e,t,mt,kc): lane l, half h ->
//     M = l%16  (o = mt*16 + M)
//     K = l<16 ? h : h+16                (c = kc*32 + K)
// ---------------------------------------------------------------------------
__global__ void lora_prep_weights(const float* __restrict__ Wc,
                                  const float* __restrict__ WA,
                                  const float* __restrict__ WB,
                                  unsigned short* __restrict__ wfrag) {
    int tid = blockIdx.x * 256 + threadIdx.x;
    if (tid >= EE * FRAGS_PER_E * FRAG_ELEMS) return;

    int h    = tid & 15;
    int lane = (tid >> 4) & 31;
    int frag = tid >> 9;

    int kc = frag & 1;
    int mt = (frag >> 1) & 3;
    int t  = (frag >> 3) % 9;
    int e  = frag / (9 * 4 * 2);

    int o = mt * 16 + (lane & 15);
    int K = (lane < 16) ? h : h + 16;
    int c  = kc * 32 + K;
    int kh = t / 3, kw = t % 3;

    float val = Wc[((o * CC + c) * 3 + kh) * 3 + kw];
    float s = 0.f;
#pragma unroll
    for (int r = 0; r < RR; ++r) {
        s += WB[(e * OO + o) * RR + r] *
             WA[(((e * RR + r) * CC + c) * 3 + kh) * 3 + kw];
    }
    val += 0.25f * s;           // ALPHA / RANK = 1/4
    wfrag[tid] = f32_to_bf16(val);
}

// ---------------------------------------------------------------------------
// Kernel 2: implicit GEMM conv, one (b, h, 128-wide column tile) per block.
// 8 waves: wave w -> M-tile (w%4) of O, N-tile group (w/4) of spatial cols.
// LDS holds x as bf16, [3 rows][132 cols][64 channels] (channel contiguous
// => each lane's B fragment = one 32B contiguous read).
// ---------------------------------------------------------------------------
__global__ __launch_bounds__(256)
void lora_conv_wmma(const float* __restrict__ x,
                    const int* __restrict__ sid,
                    const unsigned short* __restrict__ wfrag,
                    float* __restrict__ out) {
    __shared__ unsigned short s_x[3 * STG_PITCH * CC];   // ~50.7 KB

    const int tid = threadIdx.x;
    const int wt  = blockIdx.x;        // 0..1  (column tile)
    const int hrow = blockIdx.y;       // 0..63 (output row)
    const int b   = blockIdx.z;        // 0..31 (batch)
    const int w0  = wt * TILE_W;
    const int e   = sid[b];

    // ---- stage x (3 rows with halo, bf16, transposed to channel-last) ----
    for (int i = tid; i < 3 * CC * STG_COLS; i += 256) {
        int col = i % STG_COLS;            // consecutive threads -> consecutive w (coalesced)
        int t2  = i / STG_COLS;
        int c   = t2 % CC;
        int row = t2 / CC;                 // 0..2
        int hh  = hrow + row - 1;
        int ww  = w0 - 1 + col;
        float v = 0.f;
        if (hh >= 0 && hh < HH && ww >= 0 && ww < WW)
            v = x[(((size_t)b * CC + c) * HH + hh) * WW + ww];
        s_x[(row * STG_PITCH + col) * CC + c] = f32_to_bf16(v);
    }
    __syncthreads();

    // ---- per-wave GEMM ----
    const int wv   = tid >> 5;             // wave 0..7
    const int lane = tid & 31;
    const int mt   = wv & 3;               // O tile
    const int ng   = wv >> 2;              // N tile group (0/1)
    const int n    = lane & 15;
    const int loHi = (lane < 16) ? 0 : 16; // B-matrix K half per lane

    v8f acc[4] = {};

    const unsigned short* wbase =
        wfrag + (size_t)e * FRAGS_PER_E * FRAG_ELEMS;

    for (int t = 0; t < 9; ++t) {
        const int row = t / 3;
        const int dwc = t % 3;             // staged col shift (dw+1)
        for (int kc = 0; kc < 2; ++kc) {
            // A fragment (weights): one 32B load per lane, pre-swizzled
            const v16bf a = *(const v16bf*)(
                wbase + ((size_t)((t * 4 + mt) * 2 + kc)) * FRAG_ELEMS + lane * 16);
            const int cbase = kc * 32 + loHi;
#pragma unroll
            for (int ntl = 0; ntl < 4; ++ntl) {
                const int col = ng * 64 + ntl * 16 + n + dwc;
                const v16bf bf = *(const v16bf*)(
                    s_x + ((row * STG_PITCH + col) * CC + cbase));
                acc[ntl] = __builtin_amdgcn_wmma_f32_16x16x32_bf16(
                    false, a, false, bf, (short)0, acc[ntl], false, false);
            }
        }
    }

    // ---- epilogue: C/D layout -> NCHW f32 stores (coalesced over lanes) ----
    const int mHi = (lane < 16) ? 0 : 8;
#pragma unroll
    for (int ntl = 0; ntl < 4; ++ntl) {
        const int wcol = w0 + ng * 64 + ntl * 16 + n;
        float* op = out + (((size_t)b * OO + mt * 16 + mHi) * HH + hrow) * WW + wcol;
#pragma unroll
        for (int v = 0; v < 8; ++v)
            op[(size_t)v * HH * WW] = acc[ntl][v];
    }
}

extern "C" void kernel_launch(void* const* d_in, const int* in_sizes, int n_in,
                              void* d_out, int out_size, void* d_ws, size_t ws_size,
                              hipStream_t stream) {
    (void)in_sizes; (void)n_in; (void)out_size; (void)ws_size;
    const float* x   = (const float*)d_in[0];
    const int*   sid = (const int*)d_in[1];
    const float* Wc  = (const float*)d_in[2];
    const float* WA  = (const float*)d_in[3];
    const float* WB  = (const float*)d_in[4];

    unsigned short* wfrag = (unsigned short*)d_ws;   // 294,912 bytes used

    // 1) fold LoRA into per-subject effective weights, swizzled for WMMA
    const int prep_elems = EE * FRAGS_PER_E * FRAG_ELEMS;   // 147456
    lora_prep_weights<<<(prep_elems + 255) / 256, 256, 0, stream>>>(Wc, WA, WB, wfrag);

    // 2) implicit GEMM conv with bf16 WMMA
    dim3 grid(WW / TILE_W, HH, BB);   // (2, 64, 32)
    lora_conv_wmma<<<grid, 256, 0, stream>>>(x, sid, wfrag, (float*)d_out);
}
